// VectorQuantizer_6597069766768
// MI455X (gfx1250) — matrix operand order repaired and checked by reading, not compiled, not run on hardware
//
#include <hip/hip_runtime.h>
#include <hip/hip_bf16.h>

typedef __attribute__((ext_vector_type(16))) _Float16 v16h;
typedef __attribute__((ext_vector_type(8)))  float    v8f;
typedef __attribute__((ext_vector_type(4)))  float    v4f;
typedef __attribute__((ext_vector_type(2)))  float    v2f_;
typedef __attribute__((ext_vector_type(4)))  unsigned int v4u;
typedef __attribute__((ext_vector_type(8)))  int      v8i;
typedef __attribute__((ext_vector_type(4)))  int      v4i;

#define VQ_K   1024
#define VQ_D   64
#define VQ_N   65536           // 64 * 32 * 32
#define VQ_HW  1024            // 32*32
#define EMB_SCALE 1024.0f
#define INV_EMB_SCALE (1.0f/1024.0f)

// ---------------------------------------------------------------------------
// Kernel 0: emb (f32) -> emb_f16 (scaled by 1024) + enorm[k] = ||e_k||^2 (f32)
// ---------------------------------------------------------------------------
__global__ __launch_bounds__(64) void vq_prep_kernel(
    const float* __restrict__ emb, _Float16* __restrict__ e16,
    float* __restrict__ enorm)
{
    const int k = blockIdx.x;      // code
    const int d = threadIdx.x;     // dim
    __shared__ float red[64];
    float v = emb[k * VQ_D + d];
    e16[k * VQ_D + d] = (_Float16)(v * EMB_SCALE);
    red[d] = v * v;
    __syncthreads();
    if (d < 32) red[d] += red[d + 32];
    __syncthreads();
    if (d == 0) {
        float t = 0.f;
        #pragma unroll
        for (int i = 0; i < 32; ++i) t += red[i];
        enorm[k] = t;
    }
}

// ---------------------------------------------------------------------------
// Kernel 1: distance GEMM + argmin.
//  - x tile staged via the Tensor Data Mover: one tensor_load_to_lds per wave
//    (2D D#: tile 16(hw) x 64(d), tensor_dim0_stride=1024), wave waits only on
//    its OWN TENSORcnt -> no workgroup barrier at all.
//  - GEMM: v_wmma_f32_16x16x32_f16, K-dim 64 = 2 chained WMMAs per code tile.
//  - argmin score = ||e||^2 - 2*(x.e)  (||x||^2 is row-constant).
// ---------------------------------------------------------------------------
__global__ __launch_bounds__(256) void vq_argmin_kernel(
    const float* __restrict__ x,       // [64,64,32,32] NCHW
    const _Float16* __restrict__ e16,  // [1024,64] scaled f16
    const float* __restrict__ enorm,   // [1024]
    int* __restrict__ idx_out)         // [65536]
{
    // TDM writes each wave's tile contiguously in scan order: [d][m], m fastest.
    __shared__ float xs[8][VQ_D * 16];

    const int lane = threadIdx.x & 31;
    const int w    = __builtin_amdgcn_readfirstlane((int)(threadIdx.x >> 5));
    const int n0   = blockIdx.x * 128 + w * 16;     // first row of this wave's tile
    const int b    = n0 >> 10;                      // image (rows per image = 1024)
    const int hw0  = n0 & 1023;

    // ---- Tensor DMA descriptor (D#), packed per CDNA5 ISA Ch.8 ----
    const unsigned long long gaddr =
        (unsigned long long)(const void*)x +
        ((unsigned long long)b * (VQ_D * VQ_HW) + (unsigned)hw0) * 4ull;
    const unsigned int lds_off = (unsigned int)(uintptr_t)&xs[w][0]; // LDS byte offset

    // group0: count=1 | lds_addr | global_addr[56:0] | type=2
    v4u g0 = { 1u,
               lds_off,
               (unsigned int)(gaddr & 0xffffffffull),
               (unsigned int)((gaddr >> 32) & 0x01ffffffull) | 0x80000000u };
    // group1: wg_mask=0, data_size=2 (4B) | tensor_dim0=1024 | tensor_dim1=64 |
    //         tile_dim0=16 | tile_dim1=64, tile_dim2=0 | dim0_stride=1024
    v8i g1 = { (int)(2u << 16),        // [17:16] data_size = 4 bytes
               (int)(1024u << 16),     // [63:48] tensor_dim0 lo = 1024 (hw)
               (int)(64u << 16),       // [95:80]->[111:80] tensor_dim1 = 64 (d)
               (int)(16u << 16),       // [127:112] tile_dim0 = 16
               64,                     // [143:128] tile_dim1 = 64
               1024,                   // [191:160] tensor_dim0_stride = 1024
               0, 0 };
    v4i gz4 = { 0, 0, 0, 0 };          // 2D tensor: groups 2/3 unused
    v8i gz8 = { 0, 0, 0, 0, 0, 0, 0, 0 };
    __builtin_amdgcn_tensor_load_to_lds(g0, g1, gz4, gz4, gz8, 0);

    // Build B-independent lane coords while DMA is in flight.
    const int mrow = lane & 15;
    const int g    = lane >> 4;

    __builtin_amdgcn_s_wait_tensorcnt(0);     // our tile is in LDS
    __asm__ volatile("" ::: "memory");

    // A fragments (16-bit A 16x32 layout: lane half g holds K = 8g..8g+7 in
    // halves 0..7 and K = 16+8g..23+8g in halves 8..15), two K-chunks of 32.
    // Tile element (m,d) lives at xs[w][d*16 + m].
    v16h a0, a1;
    #pragma unroll
    for (int i = 0; i < 8; ++i) {
        a0[i]     = (_Float16)xs[w][(     8 * g + i) * 16 + mrow];
        a0[i + 8] = (_Float16)xs[w][(16 + 8 * g + i) * 16 + mrow];
        a1[i]     = (_Float16)xs[w][(32 +      8 * g + i) * 16 + mrow];
        a1[i + 8] = (_Float16)xs[w][(32 + 16 + 8 * g + i) * 16 + mrow];
    }

    float best[8];
    int   bidx[8];
    #pragma unroll
    for (int v = 0; v < 8; ++v) { best[v] = 3.4e38f; bidx[v] = 0; }

    // Sweep 64 column tiles of 16 codes
    for (int ct = 0; ct < 64; ++ct) {
        const int code = ct * 16 + mrow;           // this lane's N column
        const _Float16* bp = e16 + (size_t)code * VQ_D;
        // B 32x16 layout: lane half g holds K = 8g..8g+7 and 16+8g..23+8g -> 16B loads
        typedef __attribute__((ext_vector_type(8))) _Float16 v8h_;
        v8h_ b00 = *(const v8h_*)(bp + 8 * g), b01 = *(const v8h_*)(bp + 16 + 8 * g);
        v8h_ b10 = *(const v8h_*)(bp + 32 + 8 * g), b11 = *(const v8h_*)(bp + 48 + 8 * g);
        v16h bf0 = __builtin_shufflevector(b00, b01, 0,1,2,3,4,5,6,7,8,9,10,11,12,13,14,15);        // K-chunk 0 (d 0..31)
        v16h bf1 = __builtin_shufflevector(b10, b11, 0,1,2,3,4,5,6,7,8,9,10,11,12,13,14,15);   // K-chunk 1 (d 32..63)
        if (ct < 63)
            __builtin_prefetch(bp + 16 * VQ_D, 0, 3);  // next code tile -> global_prefetch

        v8f acc = {};
        acc = __builtin_amdgcn_wmma_f32_16x16x32_f16(
                false, a0, false, bf0, (short)0, acc, false, false);
        acc = __builtin_amdgcn_wmma_f32_16x16x32_f16(
                false, a1, false, bf1, (short)0, acc, false, false);

        const float en = enorm[code];
        #pragma unroll
        for (int v = 0; v < 8; ++v) {
            float s = en - (2.0f * INV_EMB_SCALE) * acc[v];
            if (s < best[v]) { best[v] = s; bidx[v] = code; }  // strict < keeps lowest idx
        }
    }

    // C layout: lanes 0-15 hold M = v, lanes 16-31 hold M = 8+v; reduce over the
    // 16 lanes (N dimension) inside each half. Tie-break: lower code index.
    #pragma unroll
    for (int v = 0; v < 8; ++v) {
        float s = best[v];
        int   ii = bidx[v];
        #pragma unroll
        for (int msk = 8; msk >= 1; msk >>= 1) {
            float os = __shfl_xor(s, msk, 16);
            int   oi = __shfl_xor(ii, msk, 16);
            if (os < s || (os == s && oi < ii)) { s = os; ii = oi; }
        }
        if (mrow == 0) idx_out[n0 + 8 * g + v] = ii;
    }
}

// ---------------------------------------------------------------------------
// Kernel 2: codebook gather -> quant_out (NCHW, streaming NT stores),
// per-block MSE partials, LDS histogram -> global counts, idx as float.
// ---------------------------------------------------------------------------
__global__ __launch_bounds__(256) void vq_gather_kernel(
    const float* __restrict__ x, const float* __restrict__ emb,
    const int* __restrict__ idx, float* __restrict__ qout,
    float* __restrict__ idx_f, int* __restrict__ counts,
    float* __restrict__ partial)
{
    __shared__ int   hist[VQ_K];
    __shared__ float red[256];
    const int t = threadIdx.x;
    for (int i = t; i < VQ_K; i += 256) hist[i] = 0;
    __syncthreads();

    const int n = blockIdx.x * 256 + t;
    const int k = idx[n];
    idx_f[n] = (float)k;
    atomicAdd(&hist[k], 1);

    const int b = n >> 10, hw = n & 1023;
    const float* xp = x    + (size_t)b * (VQ_D * VQ_HW) + hw;
    float*       qp = qout + (size_t)b * (VQ_D * VQ_HW) + hw;
    const float* ep = emb  + (size_t)k * VQ_D;

    float acc = 0.f;
    #pragma unroll
    for (int d = 0; d < VQ_D; ++d) {
        float e  = ep[d];                 // L2-hot scattered read (emb = 256 KB)
        float xv = xp[(size_t)d * VQ_HW]; // coalesced over threads
        __builtin_nontemporal_store(e, &qp[(size_t)d * VQ_HW]); // streaming write
        float df = e - xv;
        acc += df * df;
    }
    red[t] = acc;
    __syncthreads();
    for (int s = 128; s > 0; s >>= 1) {
        if (t < s) red[t] += red[t + s];
        __syncthreads();
    }
    if (t == 0) partial[blockIdx.x] = red[0];
    for (int i = t; i < VQ_K; i += 256) {
        int c = hist[i];
        if (c) atomicAdd(&counts[i], c);
    }
}

// ---------------------------------------------------------------------------
// Kernel 3: one-hot encodings [N,K] f32 (268 MB) — the bandwidth dominator.
// Section base in d_out is ≡8 (mod 16), so each 4KB row is written as:
// float2 head | 255 aligned float4 NT stores | float2 tail.
// One block per row; idx[n] is block-uniform -> scalarized load.
// ---------------------------------------------------------------------------
__global__ __launch_bounds__(256) void vq_onehot_kernel(
    const int* __restrict__ idx, float* __restrict__ enc)
{
    const int n = blockIdx.x;
    const int t = threadIdx.x;
    const int k = idx[n];
    float* p = enc + (size_t)n * VQ_K;
    if (t < 255) {
        const int base = 2 + 4 * t;           // 16B-aligned interior chunk
        v4f v = {0.f, 0.f, 0.f, 0.f};
        const int r = k - base;
        if (r >= 0 && r < 4) v[r] = 1.0f;
        __builtin_nontemporal_store(v, (v4f*)(p + base));
    } else {
        v2f_ h  = {0.f, 0.f};
        v2f_ tl = {0.f, 0.f};
        if (k == 0)    h[0]  = 1.0f;
        if (k == 1)    h[1]  = 1.0f;
        if (k == 1022) tl[0] = 1.0f;
        if (k == 1023) tl[1] = 1.0f;
        __builtin_nontemporal_store(h,  (v2f_*)p);
        __builtin_nontemporal_store(tl, (v2f_*)(p + 1022));
    }
}

// ---------------------------------------------------------------------------
// Kernel 4: perplexity + loss (fixed-order reductions -> deterministic).
// ---------------------------------------------------------------------------
__global__ __launch_bounds__(1024) void vq_final_kernel(
    const int* __restrict__ counts, const float* __restrict__ partial,
    float* __restrict__ loss_out, float* __restrict__ perp_out)
{
    __shared__ float red[VQ_K];
    __shared__ float red2[256];
    const int t = threadIdx.x;
    float p = (float)counts[t] * (1.0f / (float)VQ_N);
    red[t] = p * __logf(p + 1e-10f);
    if (t < 256) red2[t] = partial[t];
    __syncthreads();
    for (int s = 512; s > 0; s >>= 1) {
        if (t < s) red[t] += red[t + s];
        __syncthreads();
    }
    for (int s = 128; s > 0; s >>= 1) {
        if (t < s) red2[t] += red2[t + s];
        __syncthreads();
    }
    if (t == 0) {
        float perp = __expf(-red[0]);
        float mse  = red2[0] * (1.0f / ((float)VQ_N * (float)VQ_D));
        // q_latent_loss + COMMIT * e_latent_loss = 1.25 * mse (same forward value)
        float loss = 1.25f * mse + 0.1f * ((float)VQ_K - perp) / (float)VQ_K;
        *loss_out = loss;
        *perp_out = perp;
    }
}

// ---------------------------------------------------------------------------
// d_out layout (floats, concat in return order):
//   [0]                      loss
//   [1 .. 4194305)           quant_out  (64*64*32*32)
//   [4194305]                perplexity
//   [4194306 .. 71303170)    encodings  (65536*1024)
//   [71303170 .. 71368706)   idx (as float)
// ---------------------------------------------------------------------------
extern "C" void kernel_launch(void* const* d_in, const int* in_sizes, int n_in,
                              void* d_out, int out_size, void* d_ws, size_t ws_size,
                              hipStream_t stream)
{
    const float* x   = (const float*)d_in[0];   // [64,64,32,32]
    const float* emb = (const float*)d_in[1];   // [1024,64]
    float* out = (float*)d_out;
    char*  ws  = (char*)d_ws;

    _Float16* e16    = (_Float16*)ws;             // 131072 B
    float*    enorm  = (float*)(ws + 131072);     //   4096 B
    int*      idxbuf = (int*)  (ws + 135168);     // 262144 B
    int*      counts = (int*)  (ws + 397312);     //   4096 B
    float*    partia = (float*)(ws + 401408);     //   1024 B

    float* loss_out = out;
    float* qout     = out + 1;
    float* perp_out = out + 1 + 4194304;
    float* enc      = out + 2 + 4194304;
    float* idx_f    = out + 2 + 4194304 + 67108864;

    (void)hipMemsetAsync(counts, 0, VQ_K * sizeof(int), stream);
    vq_prep_kernel  <<<VQ_K,     64,  0, stream>>>(emb, e16, enorm);
    vq_argmin_kernel<<<VQ_N/128, 256, 0, stream>>>(x, e16, enorm, idxbuf);
    vq_gather_kernel<<<VQ_N/256, 256, 0, stream>>>(x, emb, idxbuf, qout, idx_f,
                                                   counts, partia);
    vq_onehot_kernel<<<VQ_N,     256, 0, stream>>>(idxbuf, enc);
    vq_final_kernel <<<1,        1024, 0, stream>>>(counts, partia,
                                                    loss_out, perp_out);
}
